// AnchorModule_6957847019624
// MI455X (gfx1250) — matrix operand order repaired and checked by reading, not compiled, not run on hardware
//
#include <hip/hip_runtime.h>
#include <math.h>

// ---------------- CDNA5 WMMA plumbing (wave32, gfx1250) ----------------
typedef __attribute__((ext_vector_type(16))) _Float16 v16h;
typedef __attribute__((ext_vector_type(8)))  _Float16 v8h;
typedef __attribute__((ext_vector_type(8)))  float    v8f;
typedef __attribute__((ext_vector_type(4)))  unsigned int v4u;
typedef __attribute__((ext_vector_type(8)))  int      v8i;
typedef __attribute__((ext_vector_type(4)))  int      v4i;

#define WMMA(a, b, c) __builtin_amdgcn_wmma_f32_16x16x32_f16( \
    /*neg_a=*/false, (a), /*neg_b=*/false, (b), /*c_mod=*/(short)0, (c), \
    /*reuse_a=*/false, /*reuse_b=*/false)

#if __has_builtin(__builtin_amdgcn_tensor_load_to_lds) && \
    __has_builtin(__builtin_amdgcn_s_wait_tensorcnt)
#define HAVE_TDM 1
#else
#define HAVE_TDM 0
#endif

// Load a full 32-K A-matrix row fragment (two contiguous 8-half runs).
// ISA 7.12.2: lane<16 holds K {0..7,16..23}; lane>=16 holds K {8..15,24..31}.
__device__ __forceinline__ v16h load_a_row32(const _Float16* rowp, int hi) {
    v8h lo = *(const v8h*)(rowp + hi * 8);
    v8h h2 = *(const v8h*)(rowp + 16 + hi * 8);
    v16h r;
    #pragma unroll
    for (int j = 0; j < 8; ++j) { r[j] = lo[j]; r[j + 8] = h2[j]; }
    return r;
}

// Load a fragment-major packed B fragment: base[((frag)*32 + lane)*16 + j]
__device__ __forceinline__ v16h load_fragB(const _Float16* base, int frag, int lane) {
    const _Float16* p = base + ((long)frag * 32 + lane) * 16;
    v8h lo = *(const v8h*)p;
    v8h h2 = *(const v8h*)(p + 8);
    v16h r;
    #pragma unroll
    for (int j = 0; j < 8; ++j) { r[j] = lo[j]; r[j + 8] = h2[j]; }
    return r;
}

#if HAVE_TDM
// 1-D contiguous TDM copy global->LDS (bytes must be multiple of 8).
__device__ __forceinline__ void tdm_copy_1d(unsigned lds_off, const void* gsrc, unsigned bytes) {
    unsigned long long ga = (unsigned long long)(size_t)gsrc;
    unsigned n8 = bytes >> 3;                       // data_size=3 -> 8B units
    v4u g0 = { 1u,                                  // count=1 (valid descriptor)
               lds_off,                             // lds_addr
               (unsigned)(ga & 0xffffffffu),        // global_addr[31:0]
               (unsigned)((ga >> 32) & 0x01ffffffu) | 0x80000000u }; // addr[56:32] | type=2
    v8i g1 = { (int)0x00030000,                     // data_size=3 (8B), mask=0
               (int)((n8 & 0xffffu) << 16),         // tensor_dim0[15:0] @ bit48
               (int)(((n8 >> 16) & 0xffffu) | (1u << 16)), // tensor_dim0[31:16], tensor_dim1=1
               (int)((n8 & 0xffffu) << 16),         // tile_dim0 @ bits 127:112
               0,                                   // tile_dim1=0, tile_dim2=0
               (int)n8,                             // tensor_dim0_stride[31:0]
               0, 0 };
    v4i z = {};
#if defined(__clang_major__) && (__clang_major__ >= 23)
    v8i z8 = {};
    __builtin_amdgcn_tensor_load_to_lds(g0, g1, z, z, z8, 0);
#else
    __builtin_amdgcn_tensor_load_to_lds(g0, g1, z, z, 0);
#endif
}
#endif

// ---------------- problem constants ----------------
#define AN 225   // Z*Y*X anchors
#define NSN 8
#define HN 64
#define BN 16
#define TN 64
#define MN 1024  // B*T
#define NPT 128  // points per frame

#define OFF_AVEC 0
#define OFF_ATTN 65536              // 16*64*64
#define OFF_HN   (65536 + 1843200) // + M*A*NS
#define OFF_CN   (OFF_HN + 1024)

// ========== weight packing: row-major (K,N) f32 -> fragment-major f16 ==========
// dst[(((kt*NT)+nt)*32 + lane)*16 + j] = src[k][n], k = kt*32 + 8*(lane>>4) + (j&7) + 16*(j>>3)
__global__ void k_pack_rm(const float* __restrict__ src, _Float16* __restrict__ dst,
                          int Ksrc, int KT, int NT, int N) {
    int i = blockIdx.x * blockDim.x + threadIdx.x;
    if (i >= KT * NT * 512) return;
    int j = i & 15, lane = (i >> 4) & 31, f = i >> 9;
    int nt = f % NT, kt = f / NT;
    int k = kt * 32 + ((lane >> 4) << 3) + (j & 7) + ((j >> 3) << 4);
    int n = nt * 16 + (lane & 15);
    dst[i] = (k < Ksrc) ? (_Float16)src[k * N + n] : (_Float16)0.0f;
}

// conv OIDHW f32 -> fragment-major f16 with reordered k' = s*C + ic (channel-minor)
__global__ void k_pack_conv(const float* __restrict__ src, _Float16* __restrict__ dst,
                            int C, int S, int KT, int NT) {
    int i = blockIdx.x * blockDim.x + threadIdx.x;
    if (i >= KT * NT * 512) return;
    int j = i & 15, lane = (i >> 4) & 31, f = i >> 9;
    int nt = f % NT, kt = f / NT;
    int kp = kt * 32 + ((lane >> 4) << 3) + (j & 7) + ((j >> 3) << 4);     // k' = s*C + ic
    int s = kp / C, ic = kp % C;
    int o = nt * 16 + (lane & 15);
    dst[i] = (_Float16)src[(long)o * C * S + ic * S + s];
}

// ================= stage 1: distances + top-8 + grouping =================
__global__ __launch_bounds__(256) void k_topk_group(
    const float* __restrict__ x, const float* __restrict__ g_loc,
    _Float16* __restrict__ grouped) {
    __shared__ float sx[NPT * 5];
    int m = blockIdx.x, tid = threadIdx.x;
    for (int i = tid; i < NPT * 5; i += 256) sx[i] = x[(long)m * (NPT * 5) + i];
    __syncthreads();
    if (tid >= AN) return;
    int a = tid;
    int z = a / 25, r = a % 25, yq = r / 5, xq = r % 5;
    float g0 = g_loc[m * 2], g1 = g_loc[m * 2 + 1];
    float ax = (xq - 2) * 0.2f + g0;
    float ay = (yq - 2) * 0.2f + g1;
    float az = z * 0.22f + 0.1f;
    float an2 = ax * ax + ay * ay + az * az;

    float bd[NSN]; int bi_[NSN];
    #pragma unroll
    for (int s = 0; s < NSN; ++s) { bd[s] = 3.4e38f; bi_[s] = 0; }
    for (int n = 0; n < NPT; ++n) {
        float px = sx[n * 5 + 0], py = sx[n * 5 + 1], pz = sx[n * 5 + 2];
        float d = an2 - 2.0f * (ax * px + ay * py + az * pz)
                + (px * px + py * py + pz * pz);
        if (d < bd[NSN - 1]) {           // stable insertion == lax.top_k ties
            int p = NSN - 1;
            while (p > 0 && d < bd[p - 1]) { bd[p] = bd[p - 1]; bi_[p] = bi_[p - 1]; --p; }
            bd[p] = d; bi_[p] = n;
        }
    }
    long base = ((long)(m * AN + a)) * NSN;
    for (int s = 0; s < NSN; ++s) {
        int n = bi_[s];
        v8h row;
        row[0] = (_Float16)(sx[n * 5 + 0] - ax);
        row[1] = (_Float16)(sx[n * 5 + 1] - ay);
        row[2] = (_Float16)(sx[n * 5 + 2] - az);
        row[3] = (_Float16)sx[n * 5 + 3];
        row[4] = (_Float16)sx[n * 5 + 4];
        row[5] = row[6] = row[7] = (_Float16)0.0f;
        *(v8h*)(grouped + (base + s) * 8) = row;   // one b128 store
    }
}

// ================= stage 2: PointNet MLP + attention (WMMA) =================
// one wave per 16-row tile (= 2 voxels); 115200 tiles = 14400 blocks x 8 waves
__global__ __launch_bounds__(256) void k_pointnet_attn(
    const _Float16* __restrict__ grouped,
    const _Float16* __restrict__ w1f, const float* __restrict__ b1,
    const _Float16* __restrict__ w2f, const float* __restrict__ b2,
    const _Float16* __restrict__ w3f, const float* __restrict__ b3,
    const float* __restrict__ attn_w, const float* __restrict__ attn_b,
    float* __restrict__ attn_out, _Float16* __restrict__ voxel_h) {
    __shared__ __attribute__((aligned(16))) _Float16 s_f1[8][16 * 16];
    __shared__ __attribute__((aligned(16))) _Float16 s_f2[8][16 * 32];
    __shared__ float s_f3[8][16 * 64];
    __shared__ float s_sm[8][16];

    int wid = threadIdx.x >> 5, lane = threadIdx.x & 31;
    int tile = blockIdx.x * 8 + wid;
    int r0 = tile * 16;
    int mrow = lane & 15, hi = lane >> 4;

    // ---- layer 1: (16x5 pad32) @ (32x16). Row data lives in K 0..7 only:
    // lanes<16 hold K 0..7 (one b128); lanes>=16 hold K 8..15 -> all zero.
    v16h a = {};
    if (hi == 0) {
        v8h lo = *(const v8h*)(grouped + (long)(r0 + mrow) * 8);
        #pragma unroll
        for (int j = 0; j < 8; ++j) a[j] = lo[j];
    }
    v16h b = load_fragB(w1f, 0, lane);
    v8f c1 = {};
    c1 = WMMA(a, b, c1);
    {
        float bv = b1[mrow];
        #pragma unroll
        for (int r = 0; r < 8; ++r) {
            float v = c1[r] + bv; v = v > 0.f ? v : 0.f;
            s_f1[wid][(r + 8 * hi) * 16 + mrow] = (_Float16)v;
        }
    }
    __syncthreads();

    // ---- layer 2: (16x16 pad32) @ (32x32); K 0..15 valid -> one ds_load_b128
    a = (v16h){};
    {
        v8h lo = *(const v8h*)&s_f1[wid][mrow * 16 + hi * 8];
        #pragma unroll
        for (int j = 0; j < 8; ++j) a[j] = lo[j];
    }
    v8f c2[2];
    #pragma unroll
    for (int nt = 0; nt < 2; ++nt) {
        v16h bb = load_fragB(w2f, nt, lane);
        v8f zz = {};
        c2[nt] = WMMA(a, bb, zz);
    }
    #pragma unroll
    for (int nt = 0; nt < 2; ++nt) {
        float bv = b2[nt * 16 + mrow];
        #pragma unroll
        for (int r = 0; r < 8; ++r) {
            float v = c2[nt][r] + bv; v = v > 0.f ? v : 0.f;
            s_f2[wid][(r + 8 * hi) * 32 + nt * 16 + mrow] = (_Float16)v;
        }
    }
    __syncthreads();

    // ---- layer 3: (16x32) @ (32x64) ----
    a = load_a_row32(&s_f2[wid][mrow * 32], hi);
    v8f c3[4];
    #pragma unroll
    for (int nt = 0; nt < 4; ++nt) {
        v16h bb = load_fragB(w3f, nt, lane);
        v8f zz = {};
        c3[nt] = WMMA(a, bb, zz);
    }
    #pragma unroll
    for (int nt = 0; nt < 4; ++nt) {
        float bv = b3[nt * 16 + mrow];
        #pragma unroll
        for (int r = 0; r < 8; ++r) {
            float v = c3[nt][r] + bv; v = v > 0.f ? v : 0.f;
            s_f3[wid][(r + 8 * hi) * 64 + nt * 16 + mrow] = v;
        }
    }
    __syncthreads();

    // ---- attention logits + softmax over NS=8 ----
    float lg = 0.0f;
    if (lane < 16) {
        for (int cc = 0; cc < 64; ++cc) lg += s_f3[wid][lane * 64 + cc] * attn_w[cc];
        lg += attn_b[0];
    }
    float mx = lg;
    mx = fmaxf(mx, __shfl_xor(mx, 1, 32));
    mx = fmaxf(mx, __shfl_xor(mx, 2, 32));
    mx = fmaxf(mx, __shfl_xor(mx, 4, 32));
    float e = expf(lg - mx);
    float sm = e;
    sm += __shfl_xor(sm, 1, 32);
    sm += __shfl_xor(sm, 2, 32);
    sm += __shfl_xor(sm, 4, 32);
    float at = e / sm;
    if (lane < 16) {
        attn_out[r0 + lane] = at;
        s_sm[wid][lane] = at;
    }
    __syncthreads();

    // ---- weighted sum -> voxel features (f16, channel-last for conv1) ----
    for (int idx = lane; idx < 128; idx += 32) {
        int v = idx >> 6, cc = idx & 63;
        float s = 0.f;
        #pragma unroll
        for (int t = 0; t < 8; ++t)
            s += s_f3[wid][(v * 8 + t) * 64 + cc] * s_sm[wid][v * 8 + t];
        voxel_h[(long)(tile * 2 + v) * 64 + cc] = (_Float16)s;
    }
}

// ================= stage 3: convs as implicit GEMM, channel-minor K ==========
// conv1: rows=M*63, N=96, K'=27*64; weight K-tiles TDM-staged to LDS
__global__ __launch_bounds__(256) void k_conv1(
    const _Float16* __restrict__ vin, const _Float16* __restrict__ wp,
    const float* __restrict__ bias, _Float16* __restrict__ vout) {
#if HAVE_TDM
    __shared__ __attribute__((aligned(16))) _Float16 s_bf[2][6 * 512]; // 6KB x2
#endif
    int wid = threadIdx.x >> 5, lane = threadIdx.x & 31;
    int rt = blockIdx.x * 8 + wid;
    int mrow = lane & 15, hi = lane >> 4;
    int row = rt * 16 + mrow;
    int m = row / 63, pix = row % 63;
    int od = pix / 9, r9 = pix % 9, oh = r9 / 3, ow = r9 % 3;
    v8f acc[6]; v8f zero = {};
    #pragma unroll
    for (int nt = 0; nt < 6; ++nt) acc[nt] = zero;

#if HAVE_TDM
    if (wid == 0 && lane == 0)
        tdm_copy_1d((unsigned)(size_t)(void*)&s_bf[0][0], wp, 6 * 512 * 2);
#endif
    for (int kt = 0; kt < 54; ++kt) {
#if HAVE_TDM
        int buf = kt & 1;
        if (wid == 0 && lane == 0) {
            if (kt + 1 < 54)
                tdm_copy_1d((unsigned)(size_t)(void*)&s_bf[buf ^ 1][0],
                            wp + (long)(kt + 1) * 6 * 512, 6 * 512 * 2);
        }
        if (wid == 0) {
            if (kt + 1 < 54) __builtin_amdgcn_s_wait_tensorcnt(1);
            else             __builtin_amdgcn_s_wait_tensorcnt(0);
        }
        __syncthreads();                         // staged frags visible
        const _Float16* bsrc = &s_bf[buf][0];
#else
        const _Float16* bsrc = wp + (long)kt * 6 * 512;
#endif
        // A fragment: K-tile = one spatial tap, half the channel block
        int s = kt >> 1;
        int kd = s / 9, rr = s % 9, kh = rr / 3, kw = rr % 3;
        const _Float16* rowp = vin +
            ((long)m * AN + (od + kd) * 25 + (oh + kh) * 5 + (ow + kw)) * 64 +
            ((kt & 1) << 5);
        v16h aa = load_a_row32(rowp, hi);
        #pragma unroll
        for (int nt = 0; nt < 6; ++nt) {
            v16h bb = load_fragB(bsrc, nt, lane);
            acc[nt] = WMMA(aa, bb, acc[nt]);
        }
#if HAVE_TDM
        __syncthreads();                         // done reading buf before reuse
#endif
    }
    #pragma unroll
    for (int nt = 0; nt < 6; ++nt) {
        int oc = nt * 16 + mrow;
        float bv = bias[oc];
        #pragma unroll
        for (int r = 0; r < 8; ++r) {
            int grow = rt * 16 + r + 8 * hi;
            int mm = grow / 63, pp = grow % 63;
            float v = acc[nt][r] + bv; v = v > 0.f ? v : 0.f;
            vout[((long)mm * 63 + pp) * 96 + oc] = (_Float16)v;   // channel-last
        }
    }
}

// conv2: rows=M*5, N=128, K'=27*96
__global__ __launch_bounds__(256) void k_conv2(
    const _Float16* __restrict__ vin, const _Float16* __restrict__ wp,
    const float* __restrict__ bias, _Float16* __restrict__ vout) {
    int wid = threadIdx.x >> 5, lane = threadIdx.x & 31;
    int rt = blockIdx.x * 8 + wid;
    int mrow = lane & 15, hi = lane >> 4;
    int row = rt * 16 + mrow;
    int m = row / 5, od = row % 5;
    v8f acc[8]; v8f zero = {};
    #pragma unroll
    for (int nt = 0; nt < 8; ++nt) acc[nt] = zero;
    for (int kt = 0; kt < 81; ++kt) {
        __builtin_prefetch(wp + (long)((kt + 1) % 81) * 8 * 512, 0, 1);
        int s = kt / 3;                          // 96 channels = 3 k-tiles per tap
        int kd = s / 9, rr = s % 9, kh = rr / 3, kw = rr % 3;
        const _Float16* rowp = vin +
            ((long)m * 63 + (od + kd) * 9 + kh * 3 + kw) * 96 + (kt % 3) * 32;
        v16h aa = load_a_row32(rowp, hi);
        #pragma unroll
        for (int nt = 0; nt < 8; ++nt) {
            v16h bb = load_fragB(wp, kt * 8 + nt, lane);
            acc[nt] = WMMA(aa, bb, acc[nt]);
        }
    }
    #pragma unroll
    for (int nt = 0; nt < 8; ++nt) {
        int oc = nt * 16 + mrow;
        float bv = bias[oc];
        #pragma unroll
        for (int r = 0; r < 8; ++r) {
            int grow = rt * 16 + r + 8 * hi;
            int mm = grow / 5, oo = grow % 5;
            float v = acc[nt][r] + bv; v = v > 0.f ? v : 0.f;
            vout[((long)mm * 5 + oo) * 128 + oc] = (_Float16)v;   // channel-last
        }
    }
}

// conv3: rows=M, N=64, K'=5*128 (no relu)
__global__ __launch_bounds__(256) void k_conv3(
    const _Float16* __restrict__ vin, const _Float16* __restrict__ wp,
    const float* __restrict__ bias, _Float16* __restrict__ xh) {
    int wid = threadIdx.x >> 5, lane = threadIdx.x & 31;
    int rt = blockIdx.x * 8 + wid;
    int mrow = lane & 15, hi = lane >> 4;
    int m = rt * 16 + mrow;
    v8f acc[4]; v8f zero = {};
    #pragma unroll
    for (int nt = 0; nt < 4; ++nt) acc[nt] = zero;
    for (int kt = 0; kt < 20; ++kt) {
        int kd = kt / 4;                         // 128 channels = 4 k-tiles per tap
        const _Float16* rowp = vin + ((long)m * 5 + kd) * 128 + (kt % 4) * 32;
        v16h aa = load_a_row32(rowp, hi);
        #pragma unroll
        for (int nt = 0; nt < 4; ++nt) {
            v16h bb = load_fragB(wp, kt * 4 + nt, lane);
            acc[nt] = WMMA(aa, bb, acc[nt]);
        }
    }
    #pragma unroll
    for (int nt = 0; nt < 4; ++nt) {
        int oc = nt * 16 + mrow;
        float bv = bias[oc];
        #pragma unroll
        for (int r = 0; r < 8; ++r) {
            int grow = rt * 16 + r + 8 * hi;
            xh[(long)grow * 64 + oc] = (_Float16)(acc[nt][r] + bv);
        }
    }
}

// ================= stage 4: LSTM (single WG, 16 waves, WMMA per step) =======
__global__ __launch_bounds__(512) void k_lstm(
    const _Float16* __restrict__ xh, const _Float16* __restrict__ wif,
    const _Float16* __restrict__ whf, const float* __restrict__ bias,
    const float* __restrict__ h0, const float* __restrict__ c0,
    float* __restrict__ out_a, float* __restrict__ out_hn, float* __restrict__ out_cn) {
    __shared__ __attribute__((aligned(16))) _Float16 s_h[BN * HN];
    __shared__ float s_c[BN * HN];
    __shared__ float s_g[BN * 256];
    int tid = threadIdx.x;
    int wid = tid >> 5, lane = tid & 31;
    int mrow = lane & 15, hi = lane >> 4;

    // this wave's 16 gate-columns of Wi/Wh, held in regs across all 64 steps
    v16h wi0 = load_fragB(wif, 0 * 16 + wid, lane);
    v16h wi1 = load_fragB(wif, 1 * 16 + wid, lane);
    v16h wh0 = load_fragB(whf, 0 * 16 + wid, lane);
    v16h wh1 = load_fragB(whf, 1 * 16 + wid, lane);

    for (int u = tid; u < BN * HN; u += 512) { s_h[u] = (_Float16)h0[u]; s_c[u] = c0[u]; }
    __syncthreads();

    for (int t = 0; t < TN; ++t) {
        const _Float16* xrow = xh + ((long)(mrow * TN + t)) * 64; // m = b*T + t
        v16h ax0 = load_a_row32(xrow, hi);
        v16h ax1 = load_a_row32(xrow + 32, hi);
        v16h ah0 = load_a_row32(&s_h[mrow * 64], hi);
        v16h ah1 = load_a_row32(&s_h[mrow * 64 + 32], hi);
        v8f acc = {};
        acc = WMMA(ax0, wi0, acc);
        acc = WMMA(ax1, wi1, acc);
        acc = WMMA(ah0, wh0, acc);
        acc = WMMA(ah1, wh1, acc);
        #pragma unroll
        for (int r = 0; r < 8; ++r)
            s_g[(r + 8 * hi) * 256 + wid * 16 + mrow] = acc[r];
        __syncthreads();

        #pragma unroll
        for (int q = 0; q < 2; ++q) {
            int u = tid * 2 + q;
            int bb = u >> 6, hh = u & 63;
            float gi = s_g[bb * 256 + hh]       + bias[hh];
            float gf = s_g[bb * 256 + 64 + hh]  + bias[64 + hh];
            float gg = s_g[bb * 256 + 128 + hh] + bias[128 + hh];
            float go = s_g[bb * 256 + 192 + hh] + bias[192 + hh];
            float si = 1.f / (1.f + expf(-gi));
            float sf = 1.f / (1.f + expf(-gf));
            float so = 1.f / (1.f + expf(-go));
            float cc = sf * s_c[u] + si * tanhf(gg);
            float hv = so * tanhf(cc);
            s_c[u] = cc;
            s_h[u] = (_Float16)hv;
            out_a[((long)(bb * TN + t)) * 64 + hh] = hv;
            if (t == TN - 1) { out_hn[u] = hv; out_cn[u] = cc; }
        }
        __syncthreads();
    }
}

// ================= host-side launch =================
extern "C" void kernel_launch(void* const* d_in, const int* in_sizes, int n_in,
                              void* d_out, int out_size, void* d_ws, size_t ws_size,
                              hipStream_t stream) {
    const float* x       = (const float*)d_in[0];
    const float* g_loc   = (const float*)d_in[1];
    const float* h0      = (const float*)d_in[2];
    const float* c0      = (const float*)d_in[3];
    const float* pn_w1   = (const float*)d_in[4];
    const float* pn_b1   = (const float*)d_in[5];
    const float* pn_w2   = (const float*)d_in[6];
    const float* pn_b2   = (const float*)d_in[7];
    const float* pn_w3   = (const float*)d_in[8];
    const float* pn_b3   = (const float*)d_in[9];
    const float* attn_w  = (const float*)d_in[10];
    const float* attn_b  = (const float*)d_in[11];
    const float* vx_w1   = (const float*)d_in[12];
    const float* vx_b1   = (const float*)d_in[13];
    const float* vx_w2   = (const float*)d_in[14];
    const float* vx_b2   = (const float*)d_in[15];
    const float* vx_w3   = (const float*)d_in[16];
    const float* vx_b3   = (const float*)d_in[17];
    const float* lstm_wi = (const float*)d_in[18];
    const float* lstm_wh = (const float*)d_in[19];
    const float* lstm_b  = (const float*)d_in[20];

    float* out = (float*)d_out;

    // workspace carve (~74 MB total, L2-resident)
    char* p = (char*)d_ws;
    auto alloc = [&](size_t bytes) { void* r = (void*)p; p += (bytes + 255) & ~(size_t)255; return r; };
    _Float16* grouped = (_Float16*)alloc((size_t)1843200 * 8 * 2);
    _Float16* voxel_h = (_Float16*)alloc((size_t)230400 * 64 * 2);   // channel-last
    _Float16* v1h     = (_Float16*)alloc((size_t)1024 * 63 * 96 * 2); // channel-last
    _Float16* v2h     = (_Float16*)alloc((size_t)1024 * 5 * 128 * 2); // channel-last
    _Float16* xh      = (_Float16*)alloc((size_t)1024 * 64 * 2);
    _Float16* w1f     = (_Float16*)alloc((size_t)1 * 1 * 512 * 2);
    _Float16* w2f     = (_Float16*)alloc((size_t)1 * 2 * 512 * 2);
    _Float16* w3f     = (_Float16*)alloc((size_t)1 * 4 * 512 * 2);
    _Float16* wif     = (_Float16*)alloc((size_t)2 * 16 * 512 * 2);
    _Float16* whf     = (_Float16*)alloc((size_t)2 * 16 * 512 * 2);
    _Float16* c1f     = (_Float16*)alloc((size_t)54 * 6 * 512 * 2);
    _Float16* c2f     = (_Float16*)alloc((size_t)81 * 8 * 512 * 2);
    _Float16* c3f     = (_Float16*)alloc((size_t)20 * 4 * 512 * 2);

    auto g1 = [](long n) { return (int)((n + 255) / 256); };

    // fragment-major weight packing
    k_pack_rm<<<g1(1 * 1 * 512), 256, 0, stream>>>(pn_w1, w1f, 5, 1, 1, 16);
    k_pack_rm<<<g1(1 * 2 * 512), 256, 0, stream>>>(pn_w2, w2f, 16, 1, 2, 32);
    k_pack_rm<<<g1(1 * 4 * 512), 256, 0, stream>>>(pn_w3, w3f, 32, 1, 4, 64);
    k_pack_rm<<<g1(2 * 16 * 512), 256, 0, stream>>>(lstm_wi, wif, 64, 2, 16, 256);
    k_pack_rm<<<g1(2 * 16 * 512), 256, 0, stream>>>(lstm_wh, whf, 64, 2, 16, 256);
    k_pack_conv<<<g1(54L * 6 * 512), 256, 0, stream>>>(vx_w1, c1f, 64, 27, 54, 6);
    k_pack_conv<<<g1(81L * 8 * 512), 256, 0, stream>>>(vx_w2, c2f, 96, 27, 81, 8);
    k_pack_conv<<<g1(20L * 4 * 512), 256, 0, stream>>>(vx_w3, c3f, 128, 5, 20, 4);

    // stage 1: top-k + grouping
    k_topk_group<<<MN, 256, 0, stream>>>(x, g_loc, grouped);

    // stage 2: PointNet + attention (115200 tiles / 8 waves per block)
    k_pointnet_attn<<<14400, 256, 0, stream>>>(
        grouped, w1f, pn_b1, w2f, pn_b2, w3f, pn_b3,
        attn_w, attn_b, out + OFF_ATTN, voxel_h);

    // stage 3: convs (implicit GEMM, channel-minor K)
    k_conv1<<<504, 256, 0, stream>>>(voxel_h, c1f, vx_b1, v1h);  // 4032 tiles
    k_conv2<<<40, 256, 0, stream>>>(v1h, c2f, vx_b2, v2h);       // 320 tiles
    k_conv3<<<8, 256, 0, stream>>>(v2h, c3f, vx_b3, xh);         // 64 tiles

    // stage 4: LSTM
    k_lstm<<<1, 512, 0, stream>>>(xh, wif, whf, lstm_b, h0, c0,
                                  out + OFF_AVEC, out + OFF_HN, out + OFF_CN);
}
